// RefIndexer_2911987827145
// MI455X (gfx1250) — matrix operand order repaired and checked by reading, compiled, not run on hardware
//
#include <hip/hip_runtime.h>
#include <hip/hip_bf16.h>

// ---------------------------------------------------------------------------
// Problem constants (match reference)
// ---------------------------------------------------------------------------
#define S_      2048
#define HIDDEN_ 2048
#define QLORA_  1536
#define NH_     32
#define HD_     128
#define ROPE_   64
#define QDIM_   (NH_ * HD_)        // 4096
#define EPS_    1e-5f
#define INV_SQRT_HD 0.08838834764831845f   // 128^-0.5
#define INV_SQRT_NH 0.17677669529663687f   // 32^-0.5

typedef __attribute__((ext_vector_type(16))) __bf16 v16bf;
typedef __attribute__((ext_vector_type(8)))  float  v8f;

// ---------------------------------------------------------------------------
// WMMA fragment loaders (CDNA5 ISA 7.12.2 layouts, wave32)
//
// A (16x32 bf16, MxK): lane L -> m = L&15, half = L>>4.
//   VGPR v holds K = (v>=4 ? 16 : 0) + half*8 + (v&3)*2 + {0,1}
// B (32x16 bf16, KxN): lane L -> n = L&15, half = L>>4.
//   VGPR v holds K = (v>=4 ? 16 : 0) + half*8 + (v&3)*2 + {0,1}
// C/D (16x16 f32): lane L -> n = L&15; VGPR v -> m = v + 8*(L>>4)
//
// Both A sources (X, row-major MxK) and B sources (W, row-major NxK, so
// B[k][n] = W[n][k]) have K contiguous in memory; the compiler coalesces
// each fragment into global_load_b128 clauses (verified in round-1 disasm).
// ---------------------------------------------------------------------------
__device__ __forceinline__ v16bf load_frag_a(const __bf16* __restrict__ src,
                                             int ld, int m0, int k0) {
  const int lane = threadIdx.x & 31;
  const int m = lane & 15, half = lane >> 4;
  const __bf16* row = src + (long)(m0 + m) * ld + k0 + half * 8;
  v16bf out;
#pragma unroll
  for (int v = 0; v < 8; ++v) {
    const int k = ((v & 4) ? 16 : 0) + (v & 3) * 2;  // + half*8 folded into row
    out[2 * v]     = row[k];
    out[2 * v + 1] = row[k + 1];
  }
  return out;
}

__device__ __forceinline__ v16bf load_frag_b(const __bf16* __restrict__ w,
                                             int ld, int n0, int k0) {
  const int lane = threadIdx.x & 31;
  const int n = lane & 15, half = lane >> 4;
  const __bf16* row = w + (long)(n0 + n) * ld + k0 + half * 8;
  v16bf out;
#pragma unroll
  for (int v = 0; v < 8; ++v) {
    const int k = ((v & 4) ? 16 : 0) + (v & 3) * 2;
    out[2 * v]     = row[k];
    out[2 * v + 1] = row[k + 1];
  }
  return out;
}

// ---------------------------------------------------------------------------
// Generic C(MxN,f32) = scale * A(MxK,bf16 rm) @ W(NxK,bf16 rm)^T
// Register-blocked: one wave computes a (16*TM) x (16*TN) macro-tile.
// Per K-step: TM+TN fragment loads feed TM*TN WMMAs (4x density vs 1x1).
// Grids are sized exactly -> EXEC all-ones for every WMMA (ISA 7.12 req).
// __launch_bounds__(128, 1): lift the VGPR cap (CDNA5 allows up to 1024
// VGPRs/wave) so accumulators + fragments stay register-resident (round-2
// disasm showed scratch spills with the default occupancy-driven budget).
// ---------------------------------------------------------------------------
template <int TM, int TN>
__global__ __launch_bounds__(128, 1)
void gemm_bf16_nt(const __bf16* __restrict__ A,
                  const __bf16* __restrict__ W,
                  float* __restrict__ C,
                  int M, int N, int K, float scale) {
  const int wave = threadIdx.x >> 5;
  const int tile = blockIdx.x * (blockDim.x >> 5) + wave;
  const int strips_n = N / (16 * TN);
  const int tn0 = (tile % strips_n) * TN;
  const int tm0 = (tile / strips_n) * TM;

  v8f acc[TM][TN];
#pragma unroll
  for (int i = 0; i < TM; ++i)
#pragma unroll
    for (int j = 0; j < TN; ++j) acc[i][j] = (v8f){};

  const int lane = threadIdx.x & 31;
  const int mlane = lane & 15;

  for (int k0 = 0; k0 < K; k0 += 32) {
    // CDNA5 prefetch path (global_prefetch_b8): pull lines two K-steps ahead.
    if (k0 + 64 < K) {
      __builtin_prefetch(A + (long)(tm0 * 16 + mlane) * K + k0 + 64, 0, 1);
      __builtin_prefetch(W + (long)(tn0 * 16 + mlane) * K + k0 + 64, 0, 1);
    }
    v16bf a[TM], b[TN];
#pragma unroll
    for (int i = 0; i < TM; ++i) a[i] = load_frag_a(A, K, (tm0 + i) * 16, k0);
#pragma unroll
    for (int j = 0; j < TN; ++j) b[j] = load_frag_b(W, K, (tn0 + j) * 16, k0);
#pragma unroll
    for (int i = 0; i < TM; ++i)
#pragma unroll
      for (int j = 0; j < TN; ++j)
        acc[i][j] = __builtin_amdgcn_wmma_f32_16x16x32_bf16(
            false, a[i], false, b[j], (short)0, acc[i][j], false, false);
  }

  const int n = lane & 15, half = lane >> 4;
#pragma unroll
  for (int i = 0; i < TM; ++i)
#pragma unroll
    for (int j = 0; j < TN; ++j)
#pragma unroll
      for (int v = 0; v < 8; ++v)
        C[(long)((tm0 + i) * 16 + v + half * 8) * N + (tn0 + j) * 16 + n] =
            acc[i][j][v] * scale;
}

// ---------------------------------------------------------------------------
// fp32 -> bf16 conversion
// ---------------------------------------------------------------------------
__global__ void cvt_f32_bf16(const float* __restrict__ in,
                             __bf16* __restrict__ out, long n) {
  long i = (long)blockIdx.x * blockDim.x + threadIdx.x;
  if (i < n) out[i] = (__bf16)in[i];
}

// ---------------------------------------------------------------------------
// k row: layernorm over 128 dims, then non-interleaved RoPE on dims [0,64).
// One block of 128 threads per sequence position s.
// ---------------------------------------------------------------------------
__global__ void k_ln_rope(const float* __restrict__ kraw,
                          const float* __restrict__ gamma,
                          const float* __restrict__ beta,
                          const float* __restrict__ cosb,   // (S, 32)
                          const float* __restrict__ sinb,   // (S, 32)
                          __bf16* __restrict__ kout) {
  __shared__ float red[128];
  __shared__ float row[128];
  const int s = blockIdx.x;
  const int d = threadIdx.x;

  float x = kraw[(long)s * HD_ + d];
  red[d] = x;
  __syncthreads();
#pragma unroll
  for (int st = 64; st > 0; st >>= 1) {
    if (d < st) red[d] += red[d + st];
    __syncthreads();
  }
  const float mean = red[0] * (1.0f / HD_);
  __syncthreads();

  const float c = x - mean;
  red[d] = c * c;
  __syncthreads();
#pragma unroll
  for (int st = 64; st > 0; st >>= 1) {
    if (d < st) red[d] += red[d + st];
    __syncthreads();
  }
  const float var = red[0] * (1.0f / HD_);
  const float y = c * rsqrtf(var + EPS_) * gamma[d] + beta[d];

  row[d] = y;
  __syncthreads();

  float o;
  if (d < 32) {
    const float cc = cosb[s * 32 + d], ss = sinb[s * 32 + d];
    o = y * cc - row[d + 32] * ss;          // x1*c - x2*s
  } else if (d < 64) {
    const float cc = cosb[s * 32 + (d - 32)], ss = sinb[s * 32 + (d - 32)];
    o = row[d - 32] * ss + y * cc;          // x1*s + x2*c
  } else {
    o = y;
  }
  kout[(long)s * HD_ + d] = (__bf16)o;
}

// ---------------------------------------------------------------------------
// q RoPE: (S, NH, HD) fp32 -> bf16, RoPE on dims [0,64) of each head.
// ---------------------------------------------------------------------------
__global__ void q_rope(const float* __restrict__ qraw,
                       const float* __restrict__ cosb,
                       const float* __restrict__ sinb,
                       __bf16* __restrict__ qout) {
  const int s = blockIdx.x;
  const float* row = qraw + (long)s * QDIM_;
  __bf16* orow = qout + (long)s * QDIM_;
  for (int idx = threadIdx.x; idx < NH_ * 64; idx += blockDim.x) {
    const int h = idx >> 6, p = idx & 63;
    const int base = h * HD_;
    if (p < 32) {
      const float cc = cosb[s * 32 + p], ss = sinb[s * 32 + p];
      const float x1 = row[base + p], x2 = row[base + 32 + p];
      orow[base + p]      = (__bf16)(x1 * cc - x2 * ss);
      orow[base + 32 + p] = (__bf16)(x1 * ss + x2 * cc);
    } else {
      const int dd = 64 + 2 * (p - 32);     // pass-through dims [64,128)
      orow[base + dd]     = (__bf16)row[base + dd];
      orow[base + dd + 1] = (__bf16)row[base + dd + 1];
    }
  }
}

// ---------------------------------------------------------------------------
// index_scores[s,t] = sum_h relu(q[s,h]·k[t] * HD^-1/2) * w[s,h] + mask[s,t]
// One wave computes a 16s x 64t macro-tile. The 16 k-tile B-fragments (4 t-
// tiles x 4 K-steps, 128 VGPRs) are loaded once and stay register-resident
// across all 32 heads (__launch_bounds__(128,1) prevents scratch spills).
// Per head: all 4 q A-fragments are hoisted ahead of the 16-WMMA block so
// the loads issue as one clause and partial s_wait_loadcnt lets kk=0 WMMAs
// start while kk=2/3 fragments are still in flight (one stall per head
// instead of four, per round-3 disasm).
// ---------------------------------------------------------------------------
__global__ __launch_bounds__(128, 1)
void scores_kernel(const __bf16* __restrict__ qb,   // (S, 4096)
                   const __bf16* __restrict__ kb,   // (S, 128)
                   const float* __restrict__ wmat,  // (S, 32), pre-scaled
                   const float* __restrict__ mask,  // (S, S)
                   float* __restrict__ out) {       // (S, S)
  const int lane = threadIdx.x & 31;
  const int wave = threadIdx.x >> 5;
  const int t0 = blockIdx.x << 6;                  // 32 t-strips of 64
  const int s0 = ((blockIdx.y << 2) + wave) << 4;  // 32 blocks x 4 waves

  v16bf bfr[4][4];  // [t-tile][K-step]
#pragma unroll
  for (int tn = 0; tn < 4; ++tn)
#pragma unroll
    for (int kk = 0; kk < 4; ++kk)
      bfr[tn][kk] = load_frag_b(kb, HD_, t0 + tn * 16, kk * 32);

  const int n = lane & 15, half = lane >> 4;
  v8f total[4];
#pragma unroll
  for (int tn = 0; tn < 4; ++tn) total[tn] = (v8f){};

  for (int h = 0; h < NH_; ++h) {
    if (h + 1 < NH_)  // global_prefetch_b8 for next head's q rows
      __builtin_prefetch(qb + (long)(s0 + (lane & 15)) * QDIM_ + (h + 1) * HD_, 0, 1);

    // Hoist all 4 A-fragments: one load clause, then a dense WMMA block.
    v16bf a[4];
#pragma unroll
    for (int kk = 0; kk < 4; ++kk)
      a[kk] = load_frag_a(qb, QDIM_, s0, h * HD_ + kk * 32);

    v8f acc[4];
#pragma unroll
    for (int tn = 0; tn < 4; ++tn) acc[tn] = (v8f){};
#pragma unroll
    for (int kk = 0; kk < 4; ++kk)
#pragma unroll
      for (int tn = 0; tn < 4; ++tn)
        acc[tn] = __builtin_amdgcn_wmma_f32_16x16x32_bf16(
            false, a[kk], false, bfr[tn][kk], (short)0, acc[tn], false, false);

    float wv[8];
#pragma unroll
    for (int v = 0; v < 8; ++v)
      wv[v] = wmat[(long)(s0 + v + half * 8) * NH_ + h];
#pragma unroll
    for (int tn = 0; tn < 4; ++tn)
#pragma unroll
      for (int v = 0; v < 8; ++v) {
        float sc = acc[tn][v] * INV_SQRT_HD;
        sc = sc > 0.0f ? sc : 0.0f;
        total[tn][v] += sc * wv[v];
      }
  }
#pragma unroll
  for (int tn = 0; tn < 4; ++tn)
#pragma unroll
    for (int v = 0; v < 8; ++v) {
      const long off = (long)(s0 + v + half * 8) * S_ + t0 + tn * 16 + n;
      out[off] = total[tn][v] + mask[off];
    }
}

// ---------------------------------------------------------------------------
// Full descending argsort per row (TOPK == S). 64-bit keys in LDS:
// high 32 = order-preserving transform of fp32 score, low 32 = (S-1-index)
// so ties break toward the smaller index (jax.lax.top_k semantics).
// Bitonic sort: 2048 keys, 256 threads, 16 KB LDS.
// ---------------------------------------------------------------------------
__global__ void sort_kernel(const float* __restrict__ scores,
                            int* __restrict__ out) {
  __shared__ unsigned long long keys[S_];
  const int s = blockIdx.x;
  const float* row = scores + (long)s * S_;

  for (int i = threadIdx.x; i < S_; i += blockDim.x) {
    const unsigned u = __float_as_uint(row[i]);
    const unsigned ord = (u & 0x80000000u) ? ~u : (u | 0x80000000u);
    keys[i] = ((unsigned long long)ord << 32) | (unsigned)(S_ - 1 - i);
  }
  __syncthreads();

  for (int k = 2; k <= S_; k <<= 1) {
    for (int j = k >> 1; j > 0; j >>= 1) {
      for (int i = threadIdx.x; i < S_; i += blockDim.x) {
        const int ixj = i ^ j;
        if (ixj > i) {
          const unsigned long long a = keys[i], b = keys[ixj];
          const bool desc = ((i & k) == 0);
          if (desc ? (a < b) : (a > b)) { keys[i] = b; keys[ixj] = a; }
        }
      }
      __syncthreads();
    }
  }

  for (int i = threadIdx.x; i < S_; i += blockDim.x) {
    out[(long)s * S_ + i] = (S_ - 1) - (int)(keys[i] & 0xFFFFFFFFu);
  }
}

// ---------------------------------------------------------------------------
// Host launcher
// ---------------------------------------------------------------------------
static inline size_t align256(size_t x) { return (x + 255) & ~(size_t)255; }

extern "C" void kernel_launch(void* const* d_in, const int* in_sizes, int n_in,
                              void* d_out, int out_size, void* d_ws, size_t ws_size,
                              hipStream_t stream) {
  const float* hidden = (const float*)d_in[0];   // (S, HIDDEN)
  const float* qcomp  = (const float*)d_in[1];   // (S, QLORA)
  const float* fcos   = (const float*)d_in[2];   // (S, 32)
  const float* fsin   = (const float*)d_in[3];   // (S, 32)
  const float* mask   = (const float*)d_in[4];   // (S, S)
  const float* wq_b   = (const float*)d_in[5];   // (4096, QLORA)
  const float* wk_w   = (const float*)d_in[6];   // (128, HIDDEN)
  const float* kscale = (const float*)d_in[7];   // (128)
  const float* kbias  = (const float*)d_in[8];   // (128)
  const float* wproj  = (const float*)d_in[9];   // (32, HIDDEN)
  int* out = (int*)d_out;                        // (S, S) int32 indices

  // -------- workspace layout (~92 MB) --------
  char* ws = (char*)d_ws;
  size_t off = 0;
  __bf16* hid_bf  = (__bf16*)(ws + off); off = align256(off + (size_t)S_ * HIDDEN_ * 2);
  __bf16* qc_bf   = (__bf16*)(ws + off); off = align256(off + (size_t)S_ * QLORA_ * 2);
  __bf16* wqb_bf  = (__bf16*)(ws + off); off = align256(off + (size_t)QDIM_ * QLORA_ * 2);
  __bf16* wkw_bf  = (__bf16*)(ws + off); off = align256(off + (size_t)HD_ * HIDDEN_ * 2);
  __bf16* wpj_bf  = (__bf16*)(ws + off); off = align256(off + (size_t)NH_ * HIDDEN_ * 2);
  float*  kraw    = (float*)(ws + off);  off = align256(off + (size_t)S_ * HD_ * 4);
  __bf16* k_bf    = (__bf16*)(ws + off); off = align256(off + (size_t)S_ * HD_ * 2);
  float*  wmat    = (float*)(ws + off);  off = align256(off + (size_t)S_ * NH_ * 4);
  float*  qraw    = (float*)(ws + off);  off = align256(off + (size_t)S_ * QDIM_ * 4);
  __bf16* q_bf    = (__bf16*)(ws + off); off = align256(off + (size_t)S_ * QDIM_ * 2);
  float*  iscores = (float*)(ws + off);  off = align256(off + (size_t)S_ * S_ * 4);

  // -------- 1) fp32 -> bf16 conversions --------
  {
    long n;
    n = (long)S_ * HIDDEN_;
    cvt_f32_bf16<<<(unsigned)((n + 255) / 256), 256, 0, stream>>>(hidden, hid_bf, n);
    n = (long)S_ * QLORA_;
    cvt_f32_bf16<<<(unsigned)((n + 255) / 256), 256, 0, stream>>>(qcomp, qc_bf, n);
    n = (long)QDIM_ * QLORA_;
    cvt_f32_bf16<<<(unsigned)((n + 255) / 256), 256, 0, stream>>>(wq_b, wqb_bf, n);
    n = (long)HD_ * HIDDEN_;
    cvt_f32_bf16<<<(unsigned)((n + 255) / 256), 256, 0, stream>>>(wk_w, wkw_bf, n);
    n = (long)NH_ * HIDDEN_;
    cvt_f32_bf16<<<(unsigned)((n + 255) / 256), 256, 0, stream>>>(wproj, wpj_bf, n);
  }

  // -------- 2) k_raw = hidden @ wk_w^T : (2048 x 128), K=2048 --------
  {
    // 64x32 macro-tiles: (2048/64) * (128/32) = 32*4 = 128 waves
    gemm_bf16_nt<4, 2><<<32, 128, 0, stream>>>(hid_bf, wkw_bf, kraw,
                                               S_, HD_, HIDDEN_, 1.0f);
  }
  // -------- 3) w = hidden @ wproj^T * NH^-1/2 : (2048 x 32) --------
  {
    // 64x32 macro-tiles: (2048/64) * (32/32) = 32 waves
    gemm_bf16_nt<4, 2><<<8, 128, 0, stream>>>(hid_bf, wpj_bf, wmat,
                                              S_, NH_, HIDDEN_, INV_SQRT_NH);
  }
  // -------- 4) k layernorm + RoPE -> bf16 --------
  k_ln_rope<<<S_, 128, 0, stream>>>(kraw, kscale, kbias, fcos, fsin, k_bf);

  // -------- 5) q_raw = q_compressed @ wq_b^T : (2048 x 4096), K=1536 --------
  {
    // 64x64 macro-tiles: (2048/64) * (4096/64) = 32*64 = 2048 waves
    gemm_bf16_nt<4, 4><<<512, 128, 0, stream>>>(qc_bf, wqb_bf, qraw,
                                                S_, QDIM_, QLORA_, 1.0f);
  }
  // -------- 6) q RoPE -> bf16 --------
  q_rope<<<S_, 256, 0, stream>>>(qraw, fcos, fsin, q_bf);

  // -------- 7) fused scores + relu·w + mask (16s x 64t per wave) --------
  {
    dim3 grid(S_ / 64, S_ / 64);                        // (32 t-strips, 32 x 4 s-tiles)
    scores_kernel<<<grid, 128, 0, stream>>>(q_bf, k_bf, wmat, mask, iscores);
  }
  // -------- 8) per-row full descending argsort --------
  sort_kernel<<<S_, 256, 0, stream>>>(iscores, out);
}
